// SCAM_15152644621033
// MI455X (gfx1250) — matrix operand order repaired and checked by reading, not compiled, not run on hardware
//
#include <hip/hip_runtime.h>
#include <hip/hip_bf16.h>
#include <math.h>

typedef __attribute__((ext_vector_type(16))) __bf16 v16bf;
typedef __attribute__((ext_vector_type(8)))  float  v8f;

#define C_ 192
#define W_ 96
#define H_ 96
#define B_ 16
#define T_ 512
#define RAW_LD 196              // padded fp32 row stride (elements)
#define MT_ 6                   // 96/16 m-tiles
#define NT_ 12                  // 192/16 n-tiles
#define KT_ 6                   // 192/32 k-steps

// CDNA5 async global<->LDS ops (ASYNCcnt path), guarded so both toolchains
// and the host pass compile. Fallback is plain load/store through VGPRs.
#if defined(__has_builtin)
#  if __has_builtin(__builtin_amdgcn_global_load_async_to_lds_b32) && \
      __has_builtin(__builtin_amdgcn_global_store_async_from_lds_b32) && \
      __has_builtin(__builtin_amdgcn_s_wait_asynccnt)
#    define HAVE_ASYNC 1
#  endif
#endif
#ifndef HAVE_ASYNC
#  define HAVE_ASYNC 0
#endif

#define GLOBAL_AS __attribute__((address_space(1)))
#define LDS_AS    __attribute__((address_space(3)))

// ---------------- WMMA fragment helpers (CDNA5 16x16x32 bf16) ----------------

// A-matrix 16x32 bf16, row-major source with row stride ldk.
// Lane half kh: element j -> K = kh*16+j (contiguous 16 per half).
__device__ __forceinline__ v16bf load_a_bf(const __bf16* A, int ldk, int mBase,
                                           int kBase, int lane) {
  int m = lane & 15, kh = (lane >> 4) & 1;
  const __bf16* p = A + (mBase + m) * ldk + kBase + kh * 16;
  v16bf r;
#pragma unroll
  for (int j = 0; j < 8; ++j) { r[j] = p[j]; r[8 + j] = p[8 + j]; }
  return r;
}

// Same, converting fp32 LDS (raw activations) to bf16 on the fly.
__device__ __forceinline__ v16bf load_a_f32(const float* A, int ldk, int mBase,
                                            int kBase, int lane) {
  int m = lane & 15, kh = (lane >> 4) & 1;
  const float* p = A + (mBase + m) * ldk + kBase + kh * 16;
  v16bf r;
#pragma unroll
  for (int j = 0; j < 8; ++j) {
    r[j]     = (__bf16)p[j];
    r[8 + j] = (__bf16)p[8 + j];
  }
  return r;
}

// B-matrix 32x16 bf16 from row-major B[k][n] with row stride ldn.
// Lane half kh holds K = kBase + kh*16 + j (contiguous 16 per half), N = lane&15.
__device__ __forceinline__ v16bf load_b_strided(const __bf16* Bm, int ldn,
                                                int kBase, int nBase, int lane) {
  int n = lane & 15, kh = (lane >> 4) & 1;
  const __bf16* p = Bm + (kBase + kh * 16) * ldn + nBase + n;
  v16bf r;
#pragma unroll
  for (int j = 0; j < 16; ++j) r[j] = p[j * ldn];
  return r;
}

// Pre-packed weight fragments: [kTile][nTile][lane][16] bf16 -> one 32B load/lane.
__device__ __forceinline__ v16bf load_b_packed(const __bf16* Wf, int kT, int nT,
                                               int lane) {
  return ((const v16bf*)Wf)[((kT * NT_ + nT) * 32) + lane];
}

__device__ __forceinline__ v8f wmma_bf16(v16bf a, v16bf b, v8f c) {
  return __builtin_amdgcn_wmma_f32_16x16x32_bf16(false, a, false, b, (short)0, c,
                                                 false, false);
}

// ---------------- prep kernels ----------------

// Convert fp32 W[o][c] -> bf16 B-fragments of W^T (B[k=c][n=o]).
__global__ __launch_bounds__(256) void prep_weights(const float* __restrict__ W,
                                                    __bf16* __restrict__ Wf) {
  int tid = blockIdx.x * 256 + threadIdx.x;
  if (tid >= KT_ * NT_ * 32) return;
  int frag = tid >> 5, lane = tid & 31;
  int kT = frag / NT_, nT = frag % NT_;
  int n = lane & 15, kh = (lane >> 4) & 1;
  const float* src = W + (nT * 16 + n) * C_ + kT * 32 + kh * 16;
  v16bf v;
#pragma unroll
  for (int j = 0; j < 16; ++j) v[j] = (__bf16)src[j];
  ((v16bf*)Wf)[frag * 32 + lane] = v;
}

// temb[b][o] = sum_T mish(t[b,T]) * Wt[o][T] + bt[o]
__global__ __launch_bounds__(256) void temb_kernel(const float* __restrict__ t,
                                                   const float* __restrict__ Wt,
                                                   const float* __restrict__ bt,
                                                   float* __restrict__ temb) {
  __shared__ float mt[T_];
  int b = blockIdx.x, tid = threadIdx.x;
  for (int i = tid; i < T_; i += 256) {
    float x = t[b * T_ + i];
    float sp = (x > 20.f) ? x : log1pf(__expf(x));
    mt[i] = x * tanhf(sp);
  }
  __syncthreads();
  if (tid < C_) {
    float s = bt[tid];
    const float* wr = Wt + tid * T_;
    for (int j = 0; j < T_; ++j) s += mt[j] * wr[j];
    temb[b * C_ + tid] = s;
  }
}

// ---------------- fused main kernel: one workgroup per (b,h) ----------------
// LDS map (bytes):
//   [0      , 75264 )  rawL  fp32 [96][196]   (persistent; becomes out_l staging)
//   [75264  , 150528)  rawR  fp32 [96][196]
//   [150528 , 224256)  r2: aL/aR bf16 [96][192] -> later vvL/vvR
//   [224256 , 297984)  r3: qL bf16 + qRT bf16 -> later att f32 + P bf16 + PcT bf16
//   [297984 , 299520)  column stats (mu/rstd x2 sides)
__global__ __launch_bounds__(256) void scam_main(
    const float* __restrict__ xl, const float* __restrict__ xr,
    const float* __restrict__ lnwl, const float* __restrict__ lnbl,
    const float* __restrict__ lnwr, const float* __restrict__ lnbr,
    const __bf16* __restrict__ WfL1, const __bf16* __restrict__ WfR1,
    const __bf16* __restrict__ WfL2, const __bf16* __restrict__ WfR2,
    const float* __restrict__ bl1, const float* __restrict__ br1,
    const float* __restrict__ bl2, const float* __restrict__ br2,
    const float* __restrict__ beta, const float* __restrict__ gammav,
    const float* __restrict__ temb, float* __restrict__ outl,
    float* __restrict__ outr) {
  extern __shared__ char smem[];
  float*  rawL  = (float*)smem;
  float*  rawR  = (float*)(smem + 75264);
  __bf16* r2    = (__bf16*)(smem + 150528);
  __bf16* r3    = (__bf16*)(smem + 224256);
  float*  stats = (float*)(smem + 297984);
  __bf16* aL = r2;             __bf16* aR  = r2 + 18432;
  __bf16* qL = r3;             __bf16* qRT = r3 + 18432;
  float*  attF = (float*)r3;                       // aliases qL after barrier
  __bf16* Pb   = (__bf16*)((char*)r3 + 36864);     // aliases qRT after barrier
  __bf16* PcT  = (__bf16*)((char*)r3 + 55296);

  const int tid = threadIdx.x, lane = tid & 31, wave = tid >> 5;
  const int bI = blockIdx.x / H_, hI = blockIdx.x % H_;
  const float scale = 0.07216878364870323f;  // 192^-0.5

  // ---- load tiles (transpose [c][w] -> [w][c]), coalesced in w ----
  const size_t gbase = ((size_t)bI * C_ * H_ + hI) * W_;
#if HAVE_ASYNC
  for (int i = tid; i < C_ * W_; i += 256) {
    int c = i / W_, w = i - c * W_;
    size_t g = gbase + (size_t)c * (H_ * W_) + w;
    __builtin_amdgcn_global_load_async_to_lds_b32(
        (GLOBAL_AS int*)(xl + g),
        (LDS_AS int*)(rawL + w * RAW_LD + c), 0, 0);
    __builtin_amdgcn_global_load_async_to_lds_b32(
        (GLOBAL_AS int*)(xr + g),
        (LDS_AS int*)(rawR + w * RAW_LD + c), 0, 0);
  }
  __builtin_amdgcn_s_wait_asynccnt(0);
#else
  for (int i = tid; i < C_ * W_; i += 256) {
    int c = i / W_, w = i - c * W_;
    size_t g = gbase + (size_t)c * (H_ * W_) + w;
    rawL[w * RAW_LD + c] = xl[g];
    rawR[w * RAW_LD + c] = xr[g];
  }
#endif
  __syncthreads();

  // ---- per-position LN stats over C ----
  if (tid < 2 * W_) {
    int side = tid >= W_;
    int w = side ? tid - W_ : tid;
    const float* raw = side ? rawR : rawL;
    float s = 0.f, q = 0.f;
    for (int c = 0; c < C_; ++c) {
      float v = raw[w * RAW_LD + c];
      s += v; q += v * v;
    }
    float mu  = s * (1.0f / C_);
    float var = q * (1.0f / C_) - mu * mu;
    stats[side * 2 * W_ + w]      = mu;
    stats[side * 2 * W_ + W_ + w] = rsqrtf(var + 1e-6f);
  }
  __syncthreads();

  // ---- apply LN -> bf16 A matrices ----
  for (int i = tid; i < C_ * W_; i += 256) {
    int w = i / C_, c = i - w * C_;
    float muL = stats[w],            rsL = stats[W_ + w];
    float muR = stats[2 * W_ + w],   rsR = stats[3 * W_ + w];
    aL[w * C_ + c] = (__bf16)(((rawL[w * RAW_LD + c] - muL) * rsL) * lnwl[c] + lnbl[c]);
    aR[w * C_ + c] = (__bf16)(((rawR[w * RAW_LD + c] - muR) * rsR) * lnwr[c] + lnbr[c]);
  }
  __syncthreads();

  // ---- Q GEMMs: group by (side,nT); hoist 6 weight B-frags to registers,
  //      run 6 m-tiles (paired for two independent accumulator chains) ----
  for (int g = wave; g < 2 * NT_; g += 8) {
    int side = g / NT_, nT = g - side * NT_;
    const __bf16* A    = side ? aR : aL;
    const __bf16* Wf   = side ? WfR1 : WfL1;
    const float*  bias = side ? br1 : bl1;
    v16bf bfr[KT_];
#pragma unroll
    for (int k = 0; k < KT_; ++k) bfr[k] = load_b_packed(Wf, k, nT, lane);
    int n = lane & 15, hi = (lane >> 4) & 1;
    float bv = bias[nT * 16 + n];
#pragma unroll
    for (int mp = 0; mp < 3; ++mp) {
      int m0 = 2 * mp, m1 = 2 * mp + 1;
      v8f acc0 = {}, acc1 = {};
#pragma unroll
      for (int k = 0; k < KT_; ++k) {
        v16bf a0 = load_a_bf(A, C_, m0 * 16, k * 32, lane);
        v16bf a1 = load_a_bf(A, C_, m1 * 16, k * 32, lane);
        acc0 = wmma_bf16(a0, bfr[k], acc0);
        acc1 = wmma_bf16(a1, bfr[k], acc1);
      }
      if (!side) {
#pragma unroll
        for (int r = 0; r < 8; ++r) {
          qL[(m0 * 16 + r + 8 * hi) * C_ + nT * 16 + n] = (__bf16)(acc0[r] + bv);
          qL[(m1 * 16 + r + 8 * hi) * C_ + nT * 16 + n] = (__bf16)(acc1[r] + bv);
        }
      } else {  // store transposed -> ready as B operand of attention GEMM
#pragma unroll
        for (int r = 0; r < 8; ++r) {
          qRT[(nT * 16 + n) * W_ + m0 * 16 + r + 8 * hi] = (__bf16)(acc0[r] + bv);
          qRT[(nT * 16 + n) * W_ + m1 * 16 + r + 8 * hi] = (__bf16)(acc1[r] + bv);
        }
      }
    }
  }
  __syncthreads();

  // ---- attention: waves 0..5 each own one y-strip (nT); B-frags from qRT
  //      hoisted per k; all 6 x-tiles accumulate in VGPRs across the barrier ----
  v8f attAcc[MT_];
  int haveAtt = (wave < MT_);
  int attNT = wave;
  if (haveAtt) {
    v8f z = {};
#pragma unroll
    for (int m = 0; m < MT_; ++m) attAcc[m] = z;
#pragma unroll
    for (int k = 0; k < KT_; ++k) {
      v16bf bq = load_b_strided(qRT, W_, k * 32, attNT * 16, lane);
#pragma unroll
      for (int m = 0; m < MT_; ++m) {
        v16bf a = load_a_bf(qL, C_, m * 16, k * 32, lane);
        attAcc[m] = wmma_bf16(a, bq, attAcc[m]);
      }
    }
  }

  // ---- V GEMMs: same grouping, fp32->bf16 A on the fly, out over aL/aR ----
  __bf16* vvL = r2;
  __bf16* vvR = r2 + 18432;
  for (int g = wave; g < 2 * NT_; g += 8) {
    int side = g / NT_, nT = g - side * NT_;
    const float*  raw  = side ? rawR : rawL;
    const __bf16* Wf   = side ? WfR2 : WfL2;
    const float*  bias = side ? br2 : bl2;
    __bf16*       vv   = side ? vvR : vvL;
    v16bf bfr[KT_];
#pragma unroll
    for (int k = 0; k < KT_; ++k) bfr[k] = load_b_packed(Wf, k, nT, lane);
    int n = lane & 15, hi = (lane >> 4) & 1;
    float bv = bias[nT * 16 + n];
#pragma unroll
    for (int mp = 0; mp < 3; ++mp) {
      int m0 = 2 * mp, m1 = 2 * mp + 1;
      v8f acc0 = {}, acc1 = {};
#pragma unroll
      for (int k = 0; k < KT_; ++k) {
        v16bf a0 = load_a_f32(raw, RAW_LD, m0 * 16, k * 32, lane);
        v16bf a1 = load_a_f32(raw, RAW_LD, m1 * 16, k * 32, lane);
        acc0 = wmma_bf16(a0, bfr[k], acc0);
        acc1 = wmma_bf16(a1, bfr[k], acc1);
      }
#pragma unroll
      for (int r = 0; r < 8; ++r) {
        vv[(m0 * 16 + r + 8 * hi) * C_ + nT * 16 + n] = (__bf16)(acc0[r] + bv);
        vv[(m1 * 16 + r + 8 * hi) * C_ + nT * 16 + n] = (__bf16)(acc1[r] + bv);
      }
    }
  }
  __syncthreads();  // all qL/qRT reads done -> safe to overwrite with att

  if (haveAtt) {
    int n = lane & 15, hi = (lane >> 4) & 1;
#pragma unroll
    for (int m = 0; m < MT_; ++m)
#pragma unroll
      for (int r = 0; r < 8; ++r)
        attF[(m * 16 + r + 8 * hi) * W_ + attNT * 16 + n] = attAcc[m][r] * scale;
  }
  __syncthreads();

  // ---- softmaxes: rows (axis -1) -> Pb ; columns (axis 2) -> PcT ----
  if (tid < W_) {
    int x = tid;
    float m = -1e30f;
    for (int y = 0; y < W_; ++y) m = fmaxf(m, attF[x * W_ + y]);
    float s = 0.f;
    for (int y = 0; y < W_; ++y) s += __expf(attF[x * W_ + y] - m);
    float inv = 1.0f / s;
    for (int y = 0; y < W_; ++y)
      Pb[x * W_ + y] = (__bf16)(__expf(attF[x * W_ + y] - m) * inv);
  } else if (tid < 2 * W_) {
    int y = tid - W_;
    float m = -1e30f;
    for (int x = 0; x < W_; ++x) m = fmaxf(m, attF[x * W_ + y]);
    float s = 0.f;
    for (int x = 0; x < W_; ++x) s += __expf(attF[x * W_ + y] - m);
    float inv = 1.0f / s;
    for (int x = 0; x < W_; ++x)
      PcT[y * W_ + x] = (__bf16)(__expf(attF[x * W_ + y] - m) * inv);
  }
  __syncthreads();

  // ---- F GEMMs fused with residual + beta/gamma + temb, in-place on raw;
  //      grouped by (side,nT) with the 3 V B-frags hoisted ----
  for (int g = wave; g < 2 * NT_; g += 8) {
    int side = g / NT_, nT = g - side * NT_;
    const __bf16* A  = side ? PcT : Pb;    // F_l2r uses column-softmax^T
    const __bf16* Bm = side ? vvL : vvR;   // F_r2l uses V_r, F_l2r uses V_l
    float* raw = side ? rawR : rawL;
    const float* gm = side ? gammav : beta;
    v16bf bfr[3];
#pragma unroll
    for (int k = 0; k < 3; ++k) bfr[k] = load_b_strided(Bm, C_, k * 32, nT * 16, lane);
    int n = lane & 15, hi = (lane >> 4) & 1;
    int c = nT * 16 + n;
    float gsc = gm[c], te = temb[bI * C_ + c];
#pragma unroll
    for (int mp = 0; mp < 3; ++mp) {
      int m0 = 2 * mp, m1 = 2 * mp + 1;
      v8f acc0 = {}, acc1 = {};
#pragma unroll
      for (int k = 0; k < 3; ++k) {
        v16bf a0 = load_a_bf(A, W_, m0 * 16, k * 32, lane);
        v16bf a1 = load_a_bf(A, W_, m1 * 16, k * 32, lane);
        acc0 = wmma_bf16(a0, bfr[k], acc0);
        acc1 = wmma_bf16(a1, bfr[k], acc1);
      }
#pragma unroll
      for (int r = 0; r < 8; ++r) {
        raw[(m0 * 16 + r + 8 * hi) * RAW_LD + c] += acc0[r] * gsc + te;
        raw[(m1 * 16 + r + 8 * hi) * RAW_LD + c] += acc1[r] * gsc + te;
      }
    }
  }
  __syncthreads();

  // ---- coalesced fp32 writeout (async LDS->global when available) ----
#if HAVE_ASYNC
  for (int i = tid; i < C_ * W_; i += 256) {
    int c = i / W_, w = i - c * W_;
    size_t g = gbase + (size_t)c * (H_ * W_) + w;
    __builtin_amdgcn_global_store_async_from_lds_b32(
        (GLOBAL_AS int*)(outl + g),
        (LDS_AS int*)(rawL + w * RAW_LD + c), 0, 0);
    __builtin_amdgcn_global_store_async_from_lds_b32(
        (GLOBAL_AS int*)(outr + g),
        (LDS_AS int*)(rawR + w * RAW_LD + c), 0, 0);
  }
  __builtin_amdgcn_s_wait_asynccnt(0);
#else
  for (int i = tid; i < C_ * W_; i += 256) {
    int c = i / W_, w = i - c * W_;
    size_t g = gbase + (size_t)c * (H_ * W_) + w;
    outl[g] = rawL[w * RAW_LD + c];
    outr[g] = rawR[w * RAW_LD + c];
  }
#endif
}

// ---------------- host launcher ----------------
extern "C" void kernel_launch(void* const* d_in, const int* in_sizes, int n_in,
                              void* d_out, int out_size, void* d_ws,
                              size_t ws_size, hipStream_t stream) {
  const float* t     = (const float*)d_in[0];
  const float* xl    = (const float*)d_in[1];
  const float* xr    = (const float*)d_in[2];
  const float* lnwl  = (const float*)d_in[3];
  const float* lnbl  = (const float*)d_in[4];
  const float* lnwr  = (const float*)d_in[5];
  const float* lnbr  = (const float*)d_in[6];
  const float* Wl1   = (const float*)d_in[7];
  const float* bl1   = (const float*)d_in[8];
  const float* Wr1   = (const float*)d_in[9];
  const float* br1   = (const float*)d_in[10];
  const float* Wl2   = (const float*)d_in[11];
  const float* bl2   = (const float*)d_in[12];
  const float* Wr2   = (const float*)d_in[13];
  const float* br2   = (const float*)d_in[14];
  const float* beta  = (const float*)d_in[15];
  const float* gam   = (const float*)d_in[16];
  const float* Wt    = (const float*)d_in[17];
  const float* bt    = (const float*)d_in[18];

  float* out_l = (float*)d_out;
  float* out_r = out_l + (size_t)B_ * C_ * H_ * W_;

  char* ws = (char*)d_ws;
  __bf16* WfL1 = (__bf16*)(ws);
  __bf16* WfR1 = (__bf16*)(ws + 73728);
  __bf16* WfL2 = (__bf16*)(ws + 147456);
  __bf16* WfR2 = (__bf16*)(ws + 221184);
  float*  temb = (float*)(ws + 294912);

  prep_weights<<<9, 256, 0, stream>>>(Wl1, WfL1);
  prep_weights<<<9, 256, 0, stream>>>(Wr1, WfR1);
  prep_weights<<<9, 256, 0, stream>>>(Wl2, WfL2);
  prep_weights<<<9, 256, 0, stream>>>(Wr2, WfR2);
  temb_kernel<<<B_, 256, 0, stream>>>(t, Wt, bt, temb);

  const int smemBytes = 299520;  // fits the 320KB CDNA5 WGP LDS
  (void)hipFuncSetAttribute((const void*)scam_main,
                            hipFuncAttributeMaxDynamicSharedMemorySize,
                            smemBytes);
  scam_main<<<B_ * H_, 256, smemBytes, stream>>>(
      xl, xr, lnwl, lnbl, lnwr, lnbr, WfL1, WfR1, WfL2, WfR2, bl1, br1, bl2,
      br2, beta, gam, temb, out_l, out_r);
}
